// LSTMAutoencoder_75917841924649
// MI455X (gfx1250) — matrix operand order repaired and checked by reading, not compiled, not run on hardware
//
#include <hip/hip_runtime.h>
#include <hip/hip_bf16.h>

typedef __attribute__((ext_vector_type(16))) _Float16 v16h;
typedef __attribute__((ext_vector_type(8)))  float    v8f;

#define SEQ 128
#define NF  32
#define HD  64
#define G4  256   // 4*H
#define BT  16    // batch rows per block (one WMMA M tile)
#define NTHREADS 128
#define BATCH 4096
#define L2E 1.4426950408889634f

// ---- LDS layout (dynamic shared) ----
#define OFF_WBUF   0        // 64x256 f16 (32KB)  enc_W (32 rows used) / dec_W
#define OFF_UBUF   32768    // 64x256 f16 (32KB)  enc_U / dec_U
#define OFF_DWBUF  65536    // 64x32  f16 (4KB)   dense_W
#define OFF_BIAS   69632    // 256 f32            enc_b / dec_b
#define OFF_DBIAS  70656    // 32 f32             dense_b
#define OFF_HBUF   70784    // 2 x 16x64 f16      hidden state (ping-pong)
#define OFF_XBUF   74880    // 2 x 16x32 f16      x_t tile (ping-pong)
#define SMEM_BYTES 76928

__device__ __forceinline__ v8f splat8(float v) {
  v8f r;
#pragma unroll
  for (int i = 0; i < 8; ++i) r[i] = v;
  return r;
}

// sigmoid(z + b) with pre-scaled bias bl = -L2E*b:
//   rcp(1 + exp2_raw(fma(z,-L2E,bl)))  -> v_fmamk + v_exp + v_add + v_rcp
__device__ __forceinline__ float sigmoid_fma(float z, float bl) {
  return __builtin_amdgcn_rcpf(
      1.0f + __builtin_amdgcn_exp2f(__builtin_fmaf(z, -L2E, bl)));
}

// A fragment: 16x32 f16, row-major source, per documented CDNA5 layout:
// lane<16 -> M=lane, K base {0..7,16..23}; lane>=16 -> K base +8
__device__ __forceinline__ v16h load_A16(const _Float16* base, int lda) {
  const int lane = threadIdx.x & 31;
  const int hlf  = lane >> 4;
  const int m    = lane & 15;
  v16h A;
#pragma unroll
  for (int r = 0; r < 8; ++r) {
    const int k = ((r & 4) << 2) + ((r & 3) << 1) + (hlf << 3);
    A[2*r+0] = base[m*lda + k + 0];
    A[2*r+1] = base[m*lda + k + 1];
  }
  return A;
}

// B fragment: 32x16 f16 (K x N), row-major source; cols [n0, n0+16)
__device__ __forceinline__ v16h load_B16(const _Float16* base, int ldb, int n0) {
  const int lane = threadIdx.x & 31;
  const int hlf  = lane >> 4;
  const int n    = lane & 15;
  v16h B;
#pragma unroll
  for (int r = 0; r < 8; ++r) {
    const int k = ((r & 4) << 2) + ((r & 3) << 1) + (hlf << 3);
    B[2*r+0] = base[(k+0)*ldb + n0 + n];
    B[2*r+1] = base[(k+1)*ldb + n0 + n];
  }
  return B;
}

__device__ __forceinline__ v8f wmma_f16(v16h a, v16h b, v8f c) {
  return __builtin_amdgcn_wmma_f32_16x16x32_f16(
      false, a, false, b, (short)0, c, false, false);
}

// stage fp32 global weights into LDS as f16, float4-vectorized
__device__ __forceinline__ void stage_f16(_Float16* dst, const float* src,
                                          int nelem, int tid) {
  for (int base = tid * 4; base < nelem; base += NTHREADS * 4) {
    const float4 v = *(const float4*)(src + base);
    dst[base+0] = (_Float16)v.x; dst[base+1] = (_Float16)v.y;
    dst[base+2] = (_Float16)v.z; dst[base+3] = (_Float16)v.w;
  }
}

__global__ __launch_bounds__(NTHREADS)
void lstm_ae_fused_kernel(const float* __restrict__ x,
                          const float* __restrict__ encW, const float* __restrict__ encU,
                          const float* __restrict__ encB,
                          const float* __restrict__ decW, const float* __restrict__ decU,
                          const float* __restrict__ decB,
                          const float* __restrict__ dnsW, const float* __restrict__ dnsB,
                          float* __restrict__ out) {
  extern __shared__ __align__(16) char smem[];
  _Float16* wbuf  = (_Float16*)(smem + OFF_WBUF);
  _Float16* ubuf  = (_Float16*)(smem + OFF_UBUF);
  _Float16* dwbuf = (_Float16*)(smem + OFF_DWBUF);
  float*    bias  = (float*)   (smem + OFF_BIAS);
  float*    dbias = (float*)   (smem + OFF_DBIAS);
  _Float16* hbuf  = (_Float16*)(smem + OFF_HBUF);   // [2][16*64]
  _Float16* xbuf  = (_Float16*)(smem + OFF_XBUF);   // [2][16*32]

  const int tid  = threadIdx.x;
  const int wave = tid >> 5;            // wave w owns cols [w*16, w*16+16) of each gate
  const int lane = tid & 31;
  const int hlf  = lane >> 4;
  const int n    = lane & 15;
  const int b0   = blockIdx.x * BT;

  // ---------------- Phase 0: stage encoder weights + x_0 ----------------
  stage_f16(wbuf,  encW, NF * G4, tid);
  stage_f16(ubuf,  encU, HD * G4, tid);
  stage_f16(dwbuf, dnsW, HD * NF, tid);
  for (int i = tid; i < G4; i += NTHREADS) bias[i]  = encB[i];
  for (int i = tid; i < NF; i += NTHREADS) dbias[i] = dnsB[i];
  for (int i = tid; i < BT * HD; i += NTHREADS) hbuf[i] = (_Float16)0.0f; // h_{-1}=0 in buf0
  {
    const int i = tid >> 3, fq = (tid & 7) << 2;  // 16 rows x 32 cols, float4 each
    const float4 v = *(const float4*)(x + (size_t)(b0 + i) * SEQ * NF + fq);
    xbuf[i*NF+fq+0] = (_Float16)v.x; xbuf[i*NF+fq+1] = (_Float16)v.y;
    xbuf[i*NF+fq+2] = (_Float16)v.z; xbuf[i*NF+fq+3] = (_Float16)v.w;   // x_0 in buf0
  }
  __syncthreads();

  // Time-invariant operands in registers: enc_U, enc_W B-fragments
  v16h Ub[4][2], Wb[4];
  float bl[4];                                   // gate biases (pre-scaled for sigmoid)
#pragma unroll
  for (int g = 0; g < 4; ++g) {
    const int n0 = g * HD + wave * 16;
#pragma unroll
    for (int ks = 0; ks < 2; ++ks)
      Ub[g][ks] = load_B16(ubuf + ks * 32 * G4, G4, n0);
    Wb[g] = load_B16(wbuf, G4, n0);
    const float bv = bias[n0 + n];
    bl[g] = (g == 2) ? bv : (-L2E * bv);         // relu gate keeps raw bias
  }
  const v8f zfrag = splat8(0.0f);                // shared zero C (lowers to inline 0)
  v8f cfrag = splat8(0.0f);

  // ---------------- Encoder: 128 steps, 1 barrier each (ping-pong) ----------------
  for (int t = 0; t < SEQ; ++t) {
    const int p = t & 1;
    const _Float16* hr = hbuf + p * (BT * HD);
    _Float16*       hw = hbuf + (1 - p) * (BT * HD);
    const _Float16* xr = xbuf + p * (BT * NF);
    _Float16*       xw = xbuf + (1 - p) * (BT * NF);

    // issue next x-tile load early to overlap with WMMA
    float4 xv;
    const int xi = tid >> 3, fq = (tid & 7) << 2;
    if (t + 1 < SEQ) {
      const float* src = x + (size_t)(b0 + xi) * SEQ * NF + (t + 1) * NF + fq;
      xv = *(const float4*)src;
      if (t + 2 < SEQ) __builtin_prefetch(src + NF, 0, 1);
    }

    v16h ax  = load_A16(xr, NF);        // x_t, K=32
    v16h ah0 = load_A16(hr, HD);        // h_{t-1}, K=0..31
    v16h ah1 = load_A16(hr + 32, HD);   // h_{t-1}, K=32..63
    v8f z[4];
#pragma unroll
    for (int g = 0; g < 4; ++g) {
      v8f acc = wmma_f16(ax,  Wb[g],    zfrag);
      acc     = wmma_f16(ah0, Ub[g][0], acc);
      acc     = wmma_f16(ah1, Ub[g][1], acc);
      z[g] = acc;
    }
#pragma unroll
    for (int r = 0; r < 8; ++r) {
      const float ig = sigmoid_fma(z[0][r], bl[0]);
      const float fg = sigmoid_fma(z[1][r], bl[1]);
      const float gg = fmaxf(z[2][r] + bl[2], 0.0f);     // relu(z+b)
      const float og = sigmoid_fma(z[3][r], bl[3]);
      const float c  = fg * cfrag[r] + ig * gg;
      cfrag[r] = c;
      hw[(r + 8 * hlf) * HD + wave * 16 + n] = (_Float16)(og * fmaxf(c, 0.0f));
    }
    if (t + 1 < SEQ) {
      xw[xi*NF+fq+0] = (_Float16)xv.x; xw[xi*NF+fq+1] = (_Float16)xv.y;
      xw[xi*NF+fq+2] = (_Float16)xv.z; xw[xi*NF+fq+3] = (_Float16)xv.w;
    }
    __syncthreads();
  }
  // encoded = enc_hs[:,-1,:] now in hbuf buffer 0 (t=127 wrote buf (1-1)=0)

  // ---------------- Swap to decoder weights ----------------
  stage_f16(wbuf, decW, HD * G4, tid);
  stage_f16(ubuf, decU, HD * G4, tid);
  for (int i = tid; i < G4; i += NTHREADS) bias[i] = decB[i];
  __syncthreads();

  // zxd = encoded @ dec_W + dec_b : time-invariant, computed ONCE
  v8f zxdl[4];   // sigmoid gates pre-scaled by -L2E; relu gate raw
  {
    v16h ae0 = load_A16(hbuf, HD);
    v16h ae1 = load_A16(hbuf + 32, HD);
#pragma unroll
    for (int g = 0; g < 4; ++g) {
      const int n0 = g * HD + wave * 16;
      v8f acc = splat8(bias[n0 + n]);
      acc = wmma_f16(ae0, load_B16(wbuf, G4, n0), acc);
      acc = wmma_f16(ae1, load_B16(wbuf + 32 * G4, G4, n0), acc);
#pragma unroll
      for (int r = 0; r < 8; ++r)
        zxdl[g][r] = (g == 2) ? acc[r] : (-L2E * acc[r]);
    }
  }
  // hoist dec_U and dense_W fragments
#pragma unroll
  for (int g = 0; g < 4; ++g)
#pragma unroll
    for (int ks = 0; ks < 2; ++ks)
      Ub[g][ks] = load_B16(ubuf + ks * 32 * G4, G4, g * HD + wave * 16);
  v16h Db0 = load_B16(dwbuf, NF, (wave & 1) * 16);           // dense_W rows 0..31
  v16h Db1 = load_B16(dwbuf + 32 * NF, NF, (wave & 1) * 16); // dense_W rows 32..63
  const float dbv = dbias[(wave & 1) * 16 + n];
  __syncthreads();
  for (int i = tid; i < BT * HD; i += NTHREADS) hbuf[i] = (_Float16)0.0f; // h_{-1}=0 buf0
  cfrag = splat8(0.0f);
  __syncthreads();

  // ---------------- Decoder + fused dense: 128 steps, 1 barrier each ----------------
  for (int t = 0; t < SEQ; ++t) {
    const int p = t & 1;
    const _Float16* hr = hbuf + p * (BT * HD);
    _Float16*       hw = hbuf + (1 - p) * (BT * HD);

    v16h ah0 = load_A16(hr, HD);
    v16h ah1 = load_A16(hr + 32, HD);
    v8f z[4];
#pragma unroll
    for (int g = 0; g < 4; ++g) {
      v8f acc = wmma_f16(ah0, Ub[g][0], zfrag);
      acc     = wmma_f16(ah1, Ub[g][1], acc);
      z[g] = acc;
    }
#pragma unroll
    for (int r = 0; r < 8; ++r) {
      const float ig = sigmoid_fma(z[0][r], zxdl[0][r]);
      const float fg = sigmoid_fma(z[1][r], zxdl[1][r]);
      const float gg = fmaxf(z[2][r] + zxdl[2][r], 0.0f);
      const float og = sigmoid_fma(z[3][r], zxdl[3][r]);
      const float c  = fg * cfrag[r] + ig * gg;
      cfrag[r] = c;
      hw[(r + 8 * hlf) * HD + wave * 16 + n] = (_Float16)(og * fmaxf(c, 0.0f));
    }
    __syncthreads();
    // fused dense on the just-written h_t (waves 0,1; N=32)
    if (wave < 2) {
      v16h dh0 = load_A16(hw, HD);
      v16h dh1 = load_A16(hw + 32, HD);
      v8f acc = wmma_f16(dh0, Db0, zfrag);
      acc     = wmma_f16(dh1, Db1, acc);
#pragma unroll
      for (int r = 0; r < 8; ++r) {
        const int m_ = r + 8 * hlf;
        out[(size_t)(b0 + m_) * SEQ * NF + (size_t)t * NF + wave * 16 + n] = acc[r] + dbv;
      }
    }
  }
}

extern "C" void kernel_launch(void* const* d_in, const int* in_sizes, int n_in,
                              void* d_out, int out_size, void* d_ws, size_t ws_size,
                              hipStream_t stream) {
  (void)in_sizes; (void)n_in; (void)out_size; (void)d_ws; (void)ws_size;
  const float* x    = (const float*)d_in[0];
  const float* encW = (const float*)d_in[1];
  const float* encU = (const float*)d_in[2];
  const float* encB = (const float*)d_in[3];
  const float* decW = (const float*)d_in[4];
  const float* decU = (const float*)d_in[5];
  const float* decB = (const float*)d_in[6];
  const float* dnsW = (const float*)d_in[7];
  const float* dnsB = (const float*)d_in[8];

  dim3 grid(BATCH / BT);   // 256 blocks, each owns 16 batch rows end-to-end
  dim3 block(NTHREADS);    // 4 waves (wave32)
  lstm_ae_fused_kernel<<<grid, block, SMEM_BYTES, stream>>>(
      x, encW, encU, encB, decW, decU, decB, dnsW, dnsB, (float*)d_out);
}
